// Patch_enc_75299366633718
// MI455X (gfx1250) — compile-verified, matrix operand order checked by reading, not run on hardware
//
#include <hip/hip_runtime.h>
#include <hip/hip_bf16.h>
#include <math.h>

// ---------------------------------------------------------------------------
// CDNA5 (gfx1250) WMMA types
// ---------------------------------------------------------------------------
typedef __bf16 bf16_t;
typedef __attribute__((ext_vector_type(16))) __bf16 v16bf;
typedef __attribute__((ext_vector_type(8)))  __bf16 v8bf;
typedef __attribute__((ext_vector_type(8)))  float  v8f;

#define N_HEADS 8
#define N_KV    2
#define HD      32

// Load a 16-element bf16 A/B fragment slice for this lane.
// p = row base + k ; kg = (lane>>4)*8  (ISA 16-bit A/B 16x32 layout)
static __device__ __forceinline__ v16bf ld_frag(const bf16_t* p, int kg)
{
    v8bf lo = *(const v8bf*)(p + kg);
    v8bf hi = *(const v8bf*)(p + 16 + kg);
    return __builtin_shufflevector(lo, hi, 0, 1, 2, 3, 4, 5, 6, 7,
                                           8, 9, 10, 11, 12, 13, 14, 15);
}

// ---------------------------------------------------------------------------
// 1) Patch embedding: x[B,L,NV] -> X/Xb[BN*P, D] = rms(patch@Wp + posemb)
//    grid = (P, B*NV), block = D (=256)
// ---------------------------------------------------------------------------
__global__ __launch_bounds__(256)
void patch_embed_kernel(const float* __restrict__ x_enc,
                        const float* __restrict__ Wp,     // [16][D]
                        const float* __restrict__ w0,     // [D]
                        float* __restrict__ X,            // [BN*P, D] f32
                        bf16_t* __restrict__ Xb,          // [BN*P, D] bf16
                        int B, int L, int NV, int P, int D)
{
    const int p  = blockIdx.x;
    const int bn = blockIdx.y;
    const int b  = bn / NV;
    const int v  = bn - b * NV;
    const int d  = threadIdx.x;

    float acc = 0.f;
#pragma unroll
    for (int t = 0; t < 16; ++t) {
        int l = p * 16 + t;
        if (l >= L) l = L - 1;                       // replication pad (edge)
        acc += x_enc[((size_t)b * L + l) * NV + v] * Wp[t * D + d];
    }
    // sinusoidal positional embedding
    const int   i2   = d & ~1;
    const float freq = __expf((float)i2 * (-9.210340371976184f / (float)D));
    const float ang  = (float)p * freq;
    acc += (d & 1) ? __cosf(ang) : __sinf(ang);

    // RMSNorm over D
    __shared__ float red[256];
    red[d] = acc * acc;
    __syncthreads();
    for (int s = 128; s > 0; s >>= 1) {
        if (d < s) red[d] += red[d + s];
        __syncthreads();
    }
    const float r = rsqrtf(red[0] / (float)D + 1e-6f);
    const float o = acc * r * w0[d];
    const size_t idx = ((size_t)bn * P + p) * D + d;
    X[idx]  = o;
    Xb[idx] = (bf16_t)o;
}

// ---------------------------------------------------------------------------
// 2) Weight convert f32[K][N] -> bf16 transposed [N][K]; grid.y = layer
// ---------------------------------------------------------------------------
__global__ __launch_bounds__(256)
void wt_transpose_kernel(const float* __restrict__ W, bf16_t* __restrict__ Wt,
                         int K, int N)
{
    const int l = blockIdx.y;
    const float* Wl  = W  + (size_t)l * K * N;
    bf16_t*      Wtl = Wt + (size_t)l * K * N;
    const int total = K * N;
    for (int idx = blockIdx.x * blockDim.x + threadIdx.x; idx < total;
         idx += gridDim.x * blockDim.x) {
        const int k = idx / N;
        const int n = idx - k * N;
        Wtl[(size_t)n * K + k] = (bf16_t)Wl[idx];
    }
}

// ---------------------------------------------------------------------------
// 3) GEMM: C[M,N] = act(A[M,K] @ W + bias).  A bf16 [M][K], W bf16 [N][K].
//    block = 256 (8 waves); wave -> 32x64 tile (2 A-frag x 4 B-frag = 8 WMMA
//    per k-step; 12 b128 loads : 8 WMMA).  Block tile = 256x64.
//    grid = (N/64, M/256).  act: 0 = none, 1 = tanh-GELU.
//    Cf (f32) and/or Cb (bf16) outputs, either may be null.
// ---------------------------------------------------------------------------
__global__ __launch_bounds__(256)
void gemm_bf16_kernel(const bf16_t* __restrict__ A,
                      const bf16_t* __restrict__ Bt,
                      const float* __restrict__ bias,
                      float* __restrict__ Cf,
                      bf16_t* __restrict__ Cb,
                      int M, int N, int K, int act)
{
    const int lane = threadIdx.x & 31;
    const int wave = threadIdx.x >> 5;
    const int m0   = lane & 15;
    const int half = lane >> 4;
    const int kg   = half * 8;              // per-lane K-group base (ISA layout)

    const int row0 = blockIdx.y * 256 + wave * 32;
    const int col0 = blockIdx.x * 64;

    const bf16_t* ap0 = A + (size_t)(row0 +  0 + m0) * K;
    const bf16_t* ap1 = A + (size_t)(row0 + 16 + m0) * K;
    const bf16_t* bp0 = Bt + (size_t)(col0 +  0 + m0) * K;
    const bf16_t* bp1 = Bt + (size_t)(col0 + 16 + m0) * K;
    const bf16_t* bp2 = Bt + (size_t)(col0 + 32 + m0) * K;
    const bf16_t* bp3 = Bt + (size_t)(col0 + 48 + m0) * K;

    v8f a00 = {}, a01 = {}, a02 = {}, a03 = {};
    v8f a10 = {}, a11 = {}, a12 = {}, a13 = {};
    for (int k = 0; k < K; k += 32) {
        const v16bf fa0 = ld_frag(ap0 + k, kg);
        const v16bf fa1 = ld_frag(ap1 + k, kg);
        const v16bf b0  = ld_frag(bp0 + k, kg);
        const v16bf b1  = ld_frag(bp1 + k, kg);
        const v16bf b2  = ld_frag(bp2 + k, kg);
        const v16bf b3  = ld_frag(bp3 + k, kg);
        a00 = __builtin_amdgcn_wmma_f32_16x16x32_bf16(false, fa0, false, b0, (short)0, a00, false, false);
        a01 = __builtin_amdgcn_wmma_f32_16x16x32_bf16(false, fa0, false, b1, (short)0, a01, false, false);
        a02 = __builtin_amdgcn_wmma_f32_16x16x32_bf16(false, fa0, false, b2, (short)0, a02, false, false);
        a03 = __builtin_amdgcn_wmma_f32_16x16x32_bf16(false, fa0, false, b3, (short)0, a03, false, false);
        a10 = __builtin_amdgcn_wmma_f32_16x16x32_bf16(false, fa1, false, b0, (short)0, a10, false, false);
        a11 = __builtin_amdgcn_wmma_f32_16x16x32_bf16(false, fa1, false, b1, (short)0, a11, false, false);
        a12 = __builtin_amdgcn_wmma_f32_16x16x32_bf16(false, fa1, false, b2, (short)0, a12, false, false);
        a13 = __builtin_amdgcn_wmma_f32_16x16x32_bf16(false, fa1, false, b3, (short)0, a13, false, false);
    }

    v8f accs[2][4] = {{a00, a01, a02, a03}, {a10, a11, a12, a13}};
#pragma unroll
    for (int t = 0; t < 2; ++t) {
#pragma unroll
        for (int j = 0; j < 4; ++j) {
            const int   n  = col0 + j * 16 + m0;
            const float bn = bias ? bias[n] : 0.f;
#pragma unroll
            for (int r = 0; r < 8; ++r) {
                const int row = row0 + t * 16 + r + half * 8;
                float v = accs[t][j][r] + bn;
                if (act == 1) {
                    const float u = 0.7978845608028654f * (v + 0.044715f * v * v * v);
                    v = 0.5f * v * (1.f + tanhf(u));
                }
                const size_t idx = (size_t)row * N + n;
                if (Cf) Cf[idx] = v;
                if (Cb) Cb[idx] = (bf16_t)v;
            }
        }
    }
}

// ---------------------------------------------------------------------------
// 4) Causal GQA flash attention with WMMA (bf16 in, bf16 out).
//    grid = (N_HEADS, BN); block = 256 (8 waves); wave owns Q-tiles qt, qt+8, ...
//    Q: [BN,P,D] bf16 (head-major cols), K/V: [BN,P,Dkv] bf16
// ---------------------------------------------------------------------------
__global__ __launch_bounds__(256)
void attn_kernel(const bf16_t* __restrict__ Q,
                 const bf16_t* __restrict__ Kb,
                 const bf16_t* __restrict__ Vb,
                 bf16_t* __restrict__ O,
                 int P, int D, int Dkv)
{
    const int h    = blockIdx.x;
    const int b    = blockIdx.y;
    const int kvh  = h >> 2;                       // n_rep = 8/2 = 4
    const int lane = threadIdx.x & 31;
    const int wave = threadIdx.x >> 5;
    const int m0   = lane & 15;
    const int half = lane >> 4;
    const int kg   = half * 8;
    const float scale = 0.17677669529663687f;      // 1/sqrt(32)

    __shared__ bf16_t Pst[8][16][32];              // per-wave P-tile staging

    const int nqt = (P + 15) / 16;                 // 17
    for (int qt = wave; qt < nqt; qt += 8) {
        // ---- Q A-fragment (16 rows x 32 dims) ----
        int q = qt * 16 + m0; if (q >= P) q = P - 1;
        const v16bf qf = ld_frag(Q + ((size_t)b * P + q) * D + h * HD, kg);

        v8f olo = {}, ohi = {};
        float mrow[8], lrow[8];
#pragma unroll
        for (int r = 0; r < 8; ++r) { mrow[r] = -3.0e38f; lrow[r] = 0.f; }

        const int qmax = qt * 16 + 15;
        for (int kb = 0; kb * 32 <= qmax; ++kb) {
            const int k0 = kb * 32;

            // ---- K B-fragments (K tile row-major == Bt layout) ----
            int key0 = k0 + m0;      if (key0 >= P) key0 = P - 1;
            int key1 = k0 + 16 + m0; if (key1 >= P) key1 = P - 1;
            const v16bf kf0 = ld_frag(Kb + ((size_t)b * P + key0) * Dkv + kvh * HD, kg);
            const v16bf kf1 = ld_frag(Kb + ((size_t)b * P + key1) * Dkv + kvh * HD, kg);

            v8f zero = {};
            v8f s0 = __builtin_amdgcn_wmma_f32_16x16x32_bf16(
                         false, qf, false, kf0, (short)0, zero, false, false);
            v8f s1 = __builtin_amdgcn_wmma_f32_16x16x32_bf16(
                         false, qf, false, kf1, (short)0, zero, false, false);

            // ---- scale, causal mask, online softmax ----
#pragma unroll
            for (int r = 0; r < 8; ++r) {
                const int qrow = qt * 16 + r + 8 * half;
                const int j0 = k0 + m0, j1 = k0 + 16 + m0;
                float a0 = s0[r] * scale, a1 = s1[r] * scale;
                if (j0 > qrow || j0 >= P || qrow >= P) a0 = -1.0e30f;
                if (j1 > qrow || j1 >= P || qrow >= P) a1 = -1.0e30f;

                float mx = fmaxf(a0, a1);
                for (int off = 1; off < 16; off <<= 1)
                    mx = fmaxf(mx, __shfl_xor(mx, off, 32));
                const float mnew = fmaxf(mrow[r], mx);
                const float corr = __expf(mrow[r] - mnew);
                const float p0 = __expf(a0 - mnew);
                const float p1 = __expf(a1 - mnew);
                float ps = p0 + p1;
                for (int off = 1; off < 16; off <<= 1)
                    ps += __shfl_xor(ps, off, 32);
                lrow[r] = lrow[r] * corr + ps;
                mrow[r] = mnew;
                olo[r] *= corr;
                ohi[r] *= corr;
                // stage probabilities: C-layout -> LDS tile [q16][k32]
                Pst[wave][r + 8 * half][m0]      = (bf16_t)p0;
                Pst[wave][r + 8 * half][16 + m0] = (bf16_t)p1;
            }
            __builtin_amdgcn_wave_barrier();
            asm volatile("s_wait_dscnt 0" ::: "memory");

            // ---- P A-fragment from LDS (ds_load_b128 pair) ----
            const v16bf pf = ld_frag(&Pst[wave][m0][0], kg);

            // ---- V B-fragments (Vt gather: dim across lanes, key across elems)
            v16bf vf0, vf1;
#pragma unroll
            for (int i = 0; i < 8; ++i) {
                int ka  = k0 + kg + i;      if (ka  >= P) ka  = P - 1;
                int kbx = k0 + 16 + kg + i; if (kbx >= P) kbx = P - 1;
                const bf16_t* va = Vb + ((size_t)b * P + ka)  * Dkv + kvh * HD;
                const bf16_t* vb = Vb + ((size_t)b * P + kbx) * Dkv + kvh * HD;
                vf0[i]     = va[m0];
                vf0[8 + i] = vb[m0];
                vf1[i]     = va[16 + m0];
                vf1[8 + i] = vb[16 + m0];
            }
            olo = __builtin_amdgcn_wmma_f32_16x16x32_bf16(
                      false, pf, false, vf0, (short)0, olo, false, false);
            ohi = __builtin_amdgcn_wmma_f32_16x16x32_bf16(
                      false, pf, false, vf1, (short)0, ohi, false, false);
            __builtin_amdgcn_wave_barrier();   // before next iter overwrites Pst
        }

        // ---- normalize + store (bf16, feeds O-projection GEMM) ----
#pragma unroll
        for (int r = 0; r < 8; ++r) {
            const int qrow = qt * 16 + r + 8 * half;
            if (qrow < P) {
                const float inv = 1.f / lrow[r];
                bf16_t* op = O + ((size_t)b * P + qrow) * D + h * HD;
                op[m0]      = (bf16_t)(olo[r] * inv);
                op[16 + m0] = (bf16_t)(ohi[r] * inv);
            }
        }
    }
}

// ---------------------------------------------------------------------------
// 5) (residual +) RMSNorm, dual f32/bf16 output.  grid = M rows, block = 256
// ---------------------------------------------------------------------------
__global__ __launch_bounds__(256)
void rms_kernel(const float* __restrict__ xin, const float* __restrict__ res,
                const float* __restrict__ w,
                float* __restrict__ out, bf16_t* __restrict__ outb, int D)
{
    const size_t row = blockIdx.x;
    const int d = threadIdx.x;
    float v = xin[row * D + d];
    if (res) v += res[row * D + d];
    __shared__ float red[256];
    red[d] = v * v;
    __syncthreads();
    for (int s = 128; s > 0; s >>= 1) {
        if (d < s) red[d] += red[d + s];
        __syncthreads();
    }
    const float o = v * rsqrtf(red[0] / (float)D + 1e-6f) * w[d];
    if (out)  out[row * D + d]  = o;
    if (outb) outb[row * D + d] = (bf16_t)o;
}

// ---------------------------------------------------------------------------
// Host orchestration
// ---------------------------------------------------------------------------
extern "C" void kernel_launch(void* const* d_in, const int* in_sizes, int n_in,
                              void* d_out, int out_size, void* d_ws, size_t ws_size,
                              hipStream_t stream)
{
    (void)in_sizes; (void)n_in; (void)out_size; (void)ws_size;
    const int B = 32, L = 4096, NV = 8, D = 256, DFF = 512, E = 4;
    const int P = (L + 16 - 16) / 16 + 1;     // 257
    const int BN = B * NV;                    // 256
    const int M = BN * P;                     // 65792 = 256 * 257
    const int Dkv = N_KV * (D / N_HEADS);     // 64

    const float* x_enc  = (const float*)d_in[0];
    const float* W_patch= (const float*)d_in[2];
    const float* rms0_w = (const float*)d_in[3];
    const float* Wq = (const float*)d_in[4];  const float* bq = (const float*)d_in[5];
    const float* Wk = (const float*)d_in[6];  const float* bk = (const float*)d_in[7];
    const float* Wv = (const float*)d_in[8];  const float* bv = (const float*)d_in[9];
    const float* Wo = (const float*)d_in[10]; const float* bo = (const float*)d_in[11];
    const float* n1_w = (const float*)d_in[12];
    const float* n2_w = (const float*)d_in[13];
    const float* Wf1 = (const float*)d_in[14]; const float* bf1 = (const float*)d_in[15];
    const float* Wf2 = (const float*)d_in[16]; const float* bf2 = (const float*)d_in[17];
    const float* nf_w = (const float*)d_in[18];

    // workspace carve-up
    char* ws = (char*)d_ws;
    size_t off = 0;
    auto carve = [&](size_t bytes) -> char* {
        char* p = ws + off;
        off += (bytes + 255) & ~(size_t)255;
        return p;
    };
    float*  X   = (float*)carve((size_t)M * D * 4);     // f32 residual stream
    float*  TB  = (float*)carve((size_t)M * D * 4);     // f32 branch output
    bf16_t* Xb  = (bf16_t*)carve((size_t)M * D * 2);    // bf16 GEMM input copy
    bf16_t* Qb  = (bf16_t*)carve((size_t)M * D * 2);
    bf16_t* Kbb = (bf16_t*)carve((size_t)M * Dkv * 2);
    bf16_t* Vbb = (bf16_t*)carve((size_t)M * Dkv * 2);
    bf16_t* ABb = (bf16_t*)carve((size_t)M * D * 2);    // attention out (bf16)
    bf16_t* Yb  = (bf16_t*)carve((size_t)M * DFF * 2);  // FFN hidden (bf16)
    bf16_t* Wqt = (bf16_t*)carve((size_t)E * D * D * 2);
    bf16_t* Wkt = (bf16_t*)carve((size_t)E * D * Dkv * 2);
    bf16_t* Wvt = (bf16_t*)carve((size_t)E * D * Dkv * 2);
    bf16_t* Wot = (bf16_t*)carve((size_t)E * D * D * 2);
    bf16_t* Wf1t= (bf16_t*)carve((size_t)E * D * DFF * 2);
    bf16_t* Wf2t= (bf16_t*)carve((size_t)E * DFF * D * 2);

    // weights -> bf16 [N][K]
    wt_transpose_kernel<<<dim3(256, E), 256, 0, stream>>>(Wq,  Wqt,  D,   D);
    wt_transpose_kernel<<<dim3(256, E), 256, 0, stream>>>(Wk,  Wkt,  D,   Dkv);
    wt_transpose_kernel<<<dim3(256, E), 256, 0, stream>>>(Wv,  Wvt,  D,   Dkv);
    wt_transpose_kernel<<<dim3(256, E), 256, 0, stream>>>(Wo,  Wot,  D,   D);
    wt_transpose_kernel<<<dim3(256, E), 256, 0, stream>>>(Wf1, Wf1t, D,   DFF);
    wt_transpose_kernel<<<dim3(256, E), 256, 0, stream>>>(Wf2, Wf2t, DFF, D);

    // patch embed + pre-encoder RMS
    patch_embed_kernel<<<dim3(P, BN), 256, 0, stream>>>(
        x_enc, W_patch, rms0_w, X, Xb, B, L, NV, P, D);

    const dim3 gD  (D   / 64, M / 256);   // (4, 257)
    const dim3 gKv (Dkv / 64, M / 256);   // (1, 257)
    const dim3 gF  (DFF / 64, M / 256);   // (8, 257)

    for (int l = 0; l < E; ++l) {
        gemm_bf16_kernel<<<gD,  256, 0, stream>>>(Xb,  Wqt  + (size_t)l*D*D,   bq  + l*D,   nullptr, Qb,  M, D,   D,   0);
        gemm_bf16_kernel<<<gKv, 256, 0, stream>>>(Xb,  Wkt  + (size_t)l*D*Dkv, bk  + l*Dkv, nullptr, Kbb, M, Dkv, D,   0);
        gemm_bf16_kernel<<<gKv, 256, 0, stream>>>(Xb,  Wvt  + (size_t)l*D*Dkv, bv  + l*Dkv, nullptr, Vbb, M, Dkv, D,   0);
        attn_kernel<<<dim3(N_HEADS, BN), 256, 0, stream>>>(Qb, Kbb, Vbb, ABb, P, D, Dkv);
        gemm_bf16_kernel<<<gD,  256, 0, stream>>>(ABb, Wot  + (size_t)l*D*D,   bo  + l*D,   TB, nullptr, M, D,   D,   0);
        rms_kernel<<<M, 256, 0, stream>>>(X, TB, n1_w + l*D, X, Xb, D);
        gemm_bf16_kernel<<<gF,  256, 0, stream>>>(Xb,  Wf1t + (size_t)l*D*DFF, bf1 + l*DFF, nullptr, Yb,  M, DFF, D,   1);
        gemm_bf16_kernel<<<gD,  256, 0, stream>>>(Yb,  Wf2t + (size_t)l*DFF*D, bf2 + l*D,   TB, nullptr, M, D,   DFF, 0);
        rms_kernel<<<M, 256, 0, stream>>>(X, TB, n2_w + l*D, X, Xb, D);
    }

    rms_kernel<<<M, 256, 0, stream>>>(X, nullptr, nf_w, (float*)d_out, nullptr, D);
}
